// TTOpticalTransformerLlamaAttention_21552145891675
// MI455X (gfx1250) — matrix operand order repaired and checked by reading, not compiled, not run on hardware
//
#include <hip/hip_runtime.h>
#include <hip/hip_bf16.h>
#include <stdint.h>

// ---------------------------------------------------------------------------
// Quantized Llama attention for gfx1250 (MI455X), wave32 + WMMA bf16.
// Pipeline:
//   1) convert x / wq / wk / wv / wo to bf16
//   2) WMMA GEMM (async-LDS double-buffered): xq = x@wq, xk = x@wk, xv = x@wv
//   3) rope + quantize:  q (uniform 256), k (log-LUT, transposed), v (log-LUT)
//   4) flash attention with WMMA QK^T and AV, online softmax,
//      qk-output / attn / av quantization applied elementwise
//   5) WMMA GEMM: out = attn_out @ wo  -> f32 d_out
// ---------------------------------------------------------------------------

typedef __attribute__((ext_vector_type(16))) __bf16          v16bf;
typedef __attribute__((ext_vector_type(16))) unsigned short  v16us;
typedef __attribute__((ext_vector_type(8)))  float           v8f;

constexpr int SEQ    = 2048;
constexpr int DMODEL = 2048;
constexpr int NH     = 32;
constexpr int NKV    = 8;
constexpr int HDIM   = 64;
constexpr int KVD    = NKV * HDIM;   // 512

__device__ __forceinline__ unsigned short f32_to_bf16(float f) {
  unsigned int u = __builtin_bit_cast(unsigned int, f);
  u += 0x7FFFu + ((u >> 16) & 1u);        // round-to-nearest-even
  return (unsigned short)(u >> 16);
}

struct UPair { uint4 a, b; };

__device__ __forceinline__ v16us ld_frag(const unsigned short* p) {
  UPair t;
  t.a = *reinterpret_cast<const uint4*>(p);
  t.b = *reinterpret_cast<const uint4*>(p + 8);
  return __builtin_bit_cast(v16us, t);
}
__device__ __forceinline__ v16us ld_frag_split(const unsigned short* p0,
                                               const unsigned short* p1) {
  UPair t;
  t.a = *reinterpret_cast<const uint4*>(p0);
  t.b = *reinterpret_cast<const uint4*>(p1);
  return __builtin_bit_cast(v16us, t);
}

__device__ __forceinline__ v8f vzero8() {
  v8f z = {0.f, 0.f, 0.f, 0.f, 0.f, 0.f, 0.f, 0.f};
  return z;
}

// D = A(16x32 bf16) * B(32x16 bf16) + C(16x16 f32)
__device__ __forceinline__ v8f wmma_bf16(v16us a, v16us b, v8f c) {
  return __builtin_amdgcn_wmma_f32_16x16x32_bf16(
      false, __builtin_bit_cast(v16bf, a),
      false, __builtin_bit_cast(v16bf, b),
      (short)0, c, false, false);
}

// ---- CDNA5 async memory->LDS (ASYNCcnt-tracked, no VGPR data path) --------
// Generic (flat) pointers to LDS carry the LDS byte offset in their low 32
// bits (ISA 10.2 aperture mapping truncates upper bits), so this yields the
// DS address VGPR the async op wants.
__device__ __forceinline__ unsigned lds_addr32(const void* p) {
  return (unsigned)(unsigned long long)p;
}
__device__ __forceinline__ void async_gl_to_lds_b128(unsigned lds,
                                                     const void* g) {
  asm volatile("global_load_async_to_lds_b128 %0, %1, off"
               :: "v"(lds), "v"((unsigned long long)g) : "memory");
}
__device__ __forceinline__ void wait_asynccnt0() {
  asm volatile("s_wait_asynccnt 0x0" ::: "memory");
}

// uniform 256-level quantizer (reference _uniform_quant, nearest rounding)
__device__ __forceinline__ float uquant(float x, float lo, float hi) {
  float step = (hi - lo) * (1.0f / 255.0f);
  float inv  = 255.0f / (hi - lo);
  float t    = (fminf(fmaxf(x, lo), hi) - lo) * inv;
  return rintf(t) * step + lo;
}

// log-LUT 256-level quantizer (reference _lut_quant_b), lut_min = 0.02004
__device__ __forceinline__ float lutquant(float b, float scale) {
  constexpr float LUTMIN   = 0.02004f;
  constexpr float STEP     = 0.01533343f;  // ln(1/0.02004)/255
  constexpr float INVSTEP  = 65.21704f;    // 255/ln(1/0.02004)
  float mag = fminf(fmaxf(fabsf(b) * (1.0f / scale), LUTMIN), 1.0f);
  float q   = __expf(rintf(__logf(mag * (1.0f / LUTMIN)) * INVSTEP) * STEP) * LUTMIN;
  return copysignf(q * scale, b);
}

// ---------------------------------------------------------------------------
// f32 -> bf16 conversion
// ---------------------------------------------------------------------------
__global__ void cvt_bf16_kernel(const float* __restrict__ src,
                                unsigned short* __restrict__ dst, int n) {
  int i = blockIdx.x * blockDim.x + threadIdx.x;
  if (i < n) dst[i] = f32_to_bf16(src[i]);
}

// ---------------------------------------------------------------------------
// WMMA GEMM: C[f32, MxN] = A[bf16, MxK row-major] * B[bf16, KxN row-major]
// 128x128x32 tiles, 256 threads = 8 waves in a 2x4 grid, each wave 64x32.
// Tiles staged with GLOBAL_LOAD_ASYNC_TO_LDS_B128 into a double buffer:
//   wait(asynccnt 0) ; barrier ; issue(next tile, other buf) ; compute ;
//   barrier ; flip
// so the async DMA of tile k+1 overlaps the 8 WMMAs of tile k.
// ---------------------------------------------------------------------------
constexpr int BM = 128, BN = 128, BK = 32;

__global__ __launch_bounds__(256) void gemm_bf16_kernel(
    const unsigned short* __restrict__ A, const unsigned short* __restrict__ B,
    float* __restrict__ C, int M, int N, int K) {
  __shared__ __align__(16) unsigned short As[2][BM * BK];   // 2 x 8 KB
  __shared__ __align__(16) unsigned short Bs[2][BK * BN];   // 2 x 8 KB

  const int tid  = threadIdx.x;
  const int lane = tid & 31;
  const int wave = tid >> 5;
  const int half = lane >> 4;
  const int ln   = lane & 15;
  const int wm   = (wave >> 2) * 64;
  const int wn   = (wave & 3) * 32;
  const int bm   = blockIdx.y * BM;
  const int bn   = blockIdx.x * BN;

  // cooperative tile-load coordinates (16 ushorts = 32B per thread per tile)
  const int ar = tid >> 1, ac = (tid & 1) * 16;
  const int br = tid >> 3, bc = (tid & 7) * 16;

  const unsigned short* gA = A + (size_t)(bm + ar) * K + ac;   // (+ k0)
  const unsigned short* gB = B + (size_t)br * N + bn + bc;     // (+ k0*N)
  unsigned aoff[2], boff[2];
  aoff[0] = lds_addr32(&As[0][ar * BK + ac]);
  aoff[1] = lds_addr32(&As[1][ar * BK + ac]);
  boff[0] = lds_addr32(&Bs[0][br * BN + bc]);
  boff[1] = lds_addr32(&Bs[1][br * BN + bc]);

  auto issue_tile = [&](int buf, int k0) {
    const unsigned short* a0 = gA + k0;
    const unsigned short* b0 = gB + (size_t)k0 * N;
    async_gl_to_lds_b128(aoff[buf],      a0);
    async_gl_to_lds_b128(aoff[buf] + 16, a0 + 8);
    async_gl_to_lds_b128(boff[buf],      b0);
    async_gl_to_lds_b128(boff[buf] + 16, b0 + 8);
  };

  v8f acc[4][2];
#pragma unroll
  for (int i = 0; i < 4; ++i)
#pragma unroll
    for (int j = 0; j < 2; ++j) acc[i][j] = vzero8();

  issue_tile(0, 0);
  int buf = 0;
  for (int k0 = 0; k0 < K; k0 += BK) {
    wait_asynccnt0();      // this wave's tile-k0 DMA done
    __syncthreads();       // everyone's tile-k0 DMA done
    if (k0 + BK < K) issue_tile(buf ^ 1, k0 + BK);

    const unsigned short* as = As[buf];
    const unsigned short* bs = Bs[buf];
    v16us bfr[2];
#pragma unroll
    for (int nf = 0; nf < 2; ++nf)             // B frag: lane = N col, 16 K rows (table order)
#pragma unroll
      for (int s = 0; s < 16; ++s)
        bfr[nf][s] = bs[((s & 7) + half * 8 + (s >> 3) * 16) * BN + wn + nf * 16 + ln];

#pragma unroll
    for (int mf = 0; mf < 4; ++mf) {           // A frag: two 8-elem runs per lane
      const unsigned short* ap = &as[(wm + mf * 16 + ln) * BK + half * 8];
      v16us af = ld_frag_split(ap, ap + 16);
#pragma unroll
      for (int nf = 0; nf < 2; ++nf)
        acc[mf][nf] = wmma_bf16(af, bfr[nf], acc[mf][nf]);
    }
    __syncthreads();       // all waves done reading buf before it is refilled
    buf ^= 1;
  }

#pragma unroll
  for (int mf = 0; mf < 4; ++mf)
#pragma unroll
    for (int nf = 0; nf < 2; ++nf)
#pragma unroll
      for (int r = 0; r < 8; ++r) {
        int row = bm + wm + mf * 16 + r + half * 8;
        int col = bn + wn + nf * 16 + ln;
        C[(size_t)row * N + col] = acc[mf][nf][r];
      }
}

// ---------------------------------------------------------------------------
// RoPE + quantize Q: xq f32 [S][H*HD] -> bf16 Q [H][S][HD], uniform quant
// ---------------------------------------------------------------------------
__global__ void rope_quant_q_kernel(const float* __restrict__ xq,
                                    const float* __restrict__ fc,
                                    const float* __restrict__ qmm,
                                    unsigned short* __restrict__ Qo) {
  int idx = blockIdx.x * blockDim.x + threadIdx.x;   // SEQ*NH*(HD/2)
  int j = idx & 31;
  int h = (idx >> 5) & (NH - 1);
  int s = idx >> 10;
  float c  = fc[(s * 32 + j) * 2 + 0];
  float si = fc[(s * 32 + j) * 2 + 1];
  size_t base = (size_t)s * DMODEL + h * HDIM + 2 * j;
  float x0 = xq[base], x1 = xq[base + 1];
  float lo = qmm[0], hi = qmm[1];
  float o0 = uquant(x0 * c - x1 * si, lo, hi);
  float o1 = uquant(x0 * si + x1 * c, lo, hi);
  size_t ob = ((size_t)h * SEQ + s) * HDIM + 2 * j;
  Qo[ob]     = f32_to_bf16(o0);
  Qo[ob + 1] = f32_to_bf16(o1);
}

// RoPE + LUT-quantize K: xk f32 [S][KVD] -> bf16 Kt [NKV][HD][S] (transposed)
__global__ void rope_quant_k_kernel(const float* __restrict__ xk,
                                    const float* __restrict__ fc,
                                    const float* __restrict__ kmm,
                                    unsigned short* __restrict__ Kt) {
  int idx = blockIdx.x * blockDim.x + threadIdx.x;   // SEQ*NKV*(HD/2)
  int j  = idx & 31;
  int kv = (idx >> 5) & (NKV - 1);
  int s  = idx >> 8;
  float c  = fc[(s * 32 + j) * 2 + 0];
  float si = fc[(s * 32 + j) * 2 + 1];
  size_t base = (size_t)s * KVD + kv * HDIM + 2 * j;
  float x0 = xk[base], x1 = xk[base + 1];
  float scale = fmaxf(fabsf(kmm[0]), fabsf(kmm[1]));
  float o0 = lutquant(x0 * c - x1 * si, scale);
  float o1 = lutquant(x0 * si + x1 * c, scale);
  Kt[((size_t)kv * HDIM + 2 * j)     * SEQ + s] = f32_to_bf16(o0);
  Kt[((size_t)kv * HDIM + 2 * j + 1) * SEQ + s] = f32_to_bf16(o1);
}

// LUT-quantize V: xv f32 [S][KVD] -> bf16 V [NKV][S][HD]
__global__ void quant_v_kernel(const float* __restrict__ xv,
                               const float* __restrict__ vmm,
                               unsigned short* __restrict__ V) {
  int idx = blockIdx.x * blockDim.x + threadIdx.x;   // SEQ*KVD
  int d  = idx & (HDIM - 1);
  int kv = (idx >> 6) & (NKV - 1);
  int s  = idx >> 9;
  float scale = fmaxf(fabsf(vmm[0]), fabsf(vmm[1]));
  float x = xv[(size_t)s * KVD + kv * HDIM + d];
  V[((size_t)kv * SEQ + s) * HDIM + d] = f32_to_bf16(lutquant(x, scale));
}

// ---------------------------------------------------------------------------
// Flash attention: one workgroup = (head, 64 q-rows), 4 waves x 16 q-rows.
// Per key tile of 32: QK^T via 4 WMMA, qk-quant + causal + online softmax,
// attn-quant, P routed via per-wave LDS to A-fragment layout, AV via 4 WMMA.
// K/V live entirely in L2 (4 MB bf16 << 192 MB) -> direct global fragment
// loads, no workgroup barriers (waves have different causal trip counts).
// ---------------------------------------------------------------------------
__global__ __launch_bounds__(128) void attn_kernel(
    const unsigned short* __restrict__ Q,    // [NH][SEQ][HD]
    const unsigned short* __restrict__ Kt,   // [NKV][HD][SEQ]
    const unsigned short* __restrict__ V,    // [NKV][SEQ][HD]
    const float* __restrict__ qkmm, const float* __restrict__ avmm,
    unsigned short* __restrict__ O) {        // [SEQ][NH*HD] bf16
  __shared__ __align__(16) unsigned short pbuf[4][16 * 32];  // per-wave P tile

  const int h    = blockIdx.x;
  const int qb   = blockIdx.y;
  const int kvh  = h >> 2;                  // N_REP = 4
  const int lane = threadIdx.x & 31;
  const int wave = threadIdx.x >> 5;
  const int half = lane >> 4;
  const int ln   = lane & 15;
  const int qrow0 = qb * 64 + wave * 16;

  const float qk_lo = qkmm[0], qk_hi = qkmm[1];
  const float av_lo = avmm[0], av_hi = avmm[1];

  // Q A-fragments (16x64 -> two 16x32 frags), quantized bf16
  v16us qa[2];
  {
    const unsigned short* qp = Q + ((size_t)h * SEQ + qrow0 + ln) * HDIM;
#pragma unroll
    for (int c = 0; c < 2; ++c) {
      const unsigned short* p = qp + c * 32 + half * 8;
      qa[c] = ld_frag_split(p, p + 16);
    }
  }

  v8f acc[4];
#pragma unroll
  for (int nf = 0; nf < 4; ++nf) acc[nf] = vzero8();
  float mrow[8], lrow[8], alpha[8];
#pragma unroll
  for (int r = 0; r < 8; ++r) { mrow[r] = -3.0e38f; lrow[r] = 0.0f; }

  const int kend = qrow0 + 16;              // causal: keys <= last q row
  for (int kb = 0; kb < kend; kb += 32) {
    // ---- S = Q K^T for 32 keys (two 16-wide N frags) ----
    v8f sfr[2] = {vzero8(), vzero8()};
#pragma unroll
    for (int n = 0; n < 2; ++n)
#pragma unroll
      for (int c = 0; c < 2; ++c) {
        v16us kf;
#pragma unroll
        for (int s = 0; s < 16; ++s)
          kf[s] = Kt[((size_t)kvh * HDIM + c * 32 + (s & 7) + half * 8 + (s >> 3) * 16) * SEQ + kb + n * 16 + ln];
        sfr[n] = wmma_bf16(qa[c], kf, sfr[n]);
      }

    // ---- qk-quant, scale, causal mask, online softmax, attn-quant ----
    float pv[2][8];
#pragma unroll
    for (int r = 0; r < 8; ++r) {
      const int qr = qrow0 + r + half * 8;
      float x0 = uquant(sfr[0][r], qk_lo, qk_hi) * 0.125f;   // 1/sqrt(64)
      float x1 = uquant(sfr[1][r], qk_lo, qk_hi) * 0.125f;
      float sv0 = (kb + ln      <= qr) ? x0 : -3.0e38f;
      float sv1 = (kb + 16 + ln <= qr) ? x1 : -3.0e38f;
      float rmax = fmaxf(sv0, sv1);
#pragma unroll
      for (int off = 1; off < 16; off <<= 1)
        rmax = fmaxf(rmax, __shfl_xor(rmax, off, 32));
      float mnew = fmaxf(mrow[r], rmax);
      float a = __expf(mrow[r] - mnew);
      mrow[r] = mnew;
      // attn probs quantized to 256 uniform levels in [0,1]
      float p0 = rintf(__expf(sv0 - mnew) * 255.0f) * (1.0f / 255.0f);
      float p1 = rintf(__expf(sv1 - mnew) * 255.0f) * (1.0f / 255.0f);
      pv[0][r] = p0;
      pv[1][r] = p1;
      float ps = p0 + p1;
#pragma unroll
      for (int off = 1; off < 16; off <<= 1)
        ps += __shfl_xor(ps, off, 32);
      lrow[r] = lrow[r] * a + ps;
      alpha[r] = a;
    }
#pragma unroll
    for (int nf = 0; nf < 4; ++nf)
#pragma unroll
      for (int r = 0; r < 8; ++r) acc[nf][r] *= alpha[r];

    // ---- P: C-fragment layout -> A-fragment layout via per-wave LDS ----
    unsigned short* pb = pbuf[wave];
#pragma unroll
    for (int n = 0; n < 2; ++n)
#pragma unroll
      for (int r = 0; r < 8; ++r)
        pb[(r + half * 8) * 32 + n * 16 + ln] = f32_to_bf16(pv[n][r]);
    asm volatile("s_wait_dscnt 0" ::: "memory");   // same-wave LDS RAW
    v16us pa = ld_frag_split(&pb[ln * 32 + half * 8],
                             &pb[ln * 32 + half * 8 + 16]);

    // ---- acc += P (16x32) * V (32xHD) ----
#pragma unroll
    for (int nf = 0; nf < 4; ++nf) {
      v16us vf;
#pragma unroll
      for (int s = 0; s < 16; ++s)
        vf[s] = V[((size_t)kvh * SEQ + kb + (s & 7) + half * 8 + (s >> 3) * 16) * HDIM + nf * 16 + ln];
      acc[nf] = wmma_bf16(pa, vf, acc[nf]);
    }
  }

  // ---- normalize, av-quant, store bf16 attn output [SEQ][NH*HD] ----
#pragma unroll
  for (int r = 0; r < 8; ++r) {
    float inv = 1.0f / lrow[r];
    const int qr = qrow0 + r + half * 8;
#pragma unroll
    for (int nf = 0; nf < 4; ++nf) {
      float val = uquant(acc[nf][r] * inv, av_lo, av_hi);
      O[(size_t)qr * (NH * HDIM) + h * HDIM + nf * 16 + ln] = f32_to_bf16(val);
    }
  }
}

// ---------------------------------------------------------------------------
extern "C" void kernel_launch(void* const* d_in, const int* in_sizes, int n_in,
                              void* d_out, int out_size, void* d_ws, size_t ws_size,
                              hipStream_t stream) {
  (void)in_sizes; (void)n_in; (void)out_size; (void)ws_size;
  const float* x    = (const float*)d_in[0];
  const float* fc   = (const float*)d_in[1];
  const float* wq   = (const float*)d_in[2];
  const float* wk   = (const float*)d_in[3];
  const float* wv   = (const float*)d_in[4];
  const float* wo   = (const float*)d_in[5];
  const float* qmm  = (const float*)d_in[6];
  const float* kmm  = (const float*)d_in[7];
  const float* qkmm = (const float*)d_in[8];
  const float* amm  = (const float*)d_in[9];   // attn range fixed [0,1]
  const float* vmm  = (const float*)d_in[10];
  const float* avmm = (const float*)d_in[11];
  (void)amm;  // q_seed (d_in[12]) unused: nearest rounding instead of RNG

  char* ws = (char*)d_ws;
  size_t off = 0;
  auto alloc = [&](size_t bytes) {
    size_t r = off; off += (bytes + 255) & ~(size_t)255; return r;
  };
  const size_t NX  = (size_t)SEQ * DMODEL;      // 4M elems
  const size_t NKW = (size_t)DMODEL * KVD;      // 1M elems
  const size_t NKX = (size_t)SEQ * KVD;         // 1M elems

  unsigned short* x_bf  = (unsigned short*)(ws + alloc(NX * 2));
  unsigned short* wq_bf = (unsigned short*)(ws + alloc(NX * 2));
  unsigned short* wk_bf = (unsigned short*)(ws + alloc(NKW * 2));
  unsigned short* wv_bf = (unsigned short*)(ws + alloc(NKW * 2));
  unsigned short* wo_bf = (unsigned short*)(ws + alloc(NX * 2));
  float* xq_f = (float*)(ws + alloc(NX * 4));
  float* xk_f = (float*)(ws + alloc(NKX * 4));
  float* xv_f = (float*)(ws + alloc(NKX * 4));
  unsigned short* q_bf  = (unsigned short*)(ws + alloc(NX * 2));
  unsigned short* kt_bf = (unsigned short*)(ws + alloc(NKX * 2));
  unsigned short* v_bf  = (unsigned short*)(ws + alloc(NKX * 2));
  unsigned short* ao_bf = x_bf;   // reuse: x_bf dead after projection GEMMs

  const int T = 256;
  cvt_bf16_kernel<<<(int)((NX  + T - 1) / T), T, 0, stream>>>(x,  x_bf,  (int)NX);
  cvt_bf16_kernel<<<(int)((NX  + T - 1) / T), T, 0, stream>>>(wq, wq_bf, (int)NX);
  cvt_bf16_kernel<<<(int)((NKW + T - 1) / T), T, 0, stream>>>(wk, wk_bf, (int)NKW);
  cvt_bf16_kernel<<<(int)((NKW + T - 1) / T), T, 0, stream>>>(wv, wv_bf, (int)NKW);
  cvt_bf16_kernel<<<(int)((NX  + T - 1) / T), T, 0, stream>>>(wo, wo_bf, (int)NX);

  dim3 gq(DMODEL / BN, SEQ / BM);
  dim3 gk(KVD / BN, SEQ / BM);
  gemm_bf16_kernel<<<gq, 256, 0, stream>>>(x_bf, wq_bf, xq_f, SEQ, DMODEL, DMODEL);
  gemm_bf16_kernel<<<gk, 256, 0, stream>>>(x_bf, wk_bf, xk_f, SEQ, KVD, DMODEL);
  gemm_bf16_kernel<<<gk, 256, 0, stream>>>(x_bf, wv_bf, xv_f, SEQ, KVD, DMODEL);

  rope_quant_q_kernel<<<(SEQ * NH  * 32) / T, T, 0, stream>>>(xq_f, fc, qmm, q_bf);
  rope_quant_k_kernel<<<(SEQ * NKV * 32) / T, T, 0, stream>>>(xk_f, fc, kmm, kt_bf);
  quant_v_kernel<<<(SEQ * KVD) / T, T, 0, stream>>>(xv_f, vmm, v_bf);

  attn_kernel<<<dim3(NH, SEQ / 64), 128, 0, stream>>>(q_bf, kt_bf, v_bf,
                                                      qkmm, avmm, ao_bf);

  gemm_bf16_kernel<<<gq, 256, 0, stream>>>(ao_bf, wo_bf, (float*)d_out,
                                           SEQ, DMODEL, DMODEL);
}
